// FlexAttnRelativeMultiHeadAttention_64304250355919
// MI455X (gfx1250) — matrix operand order repaired and checked by reading, not compiled, not run on hardware
//
#include <hip/hip_runtime.h>
#include <hip/hip_bf16.h>

// ---------------------------------------------------------------------------
// FlexAttn relative-bias MHA for MI455X (gfx1250, wave32, WMMA bf16)
// B=2, L=2048, D=1024, H=16, HD=64
// ---------------------------------------------------------------------------

#define BATCH 2
#define SEQ   2048
#define DIM   1024
#define HEADS 16
#define HDIM  64
#define ROWS  (BATCH * SEQ)      // 4096

typedef __attribute__((ext_vector_type(16))) __bf16 v16bf;
typedef __attribute__((ext_vector_type(2)))  __bf16 v2bf;
typedef __attribute__((ext_vector_type(8)))  float  v8f;

union BFrag {
    v16bf v;
    v2bf  h[8];
    unsigned int u[8];
};

__device__ __forceinline__ v2bf cvt_pk_bf16(float2 f) {
    v2bf r;
    r.x = (__bf16)f.x;     // native v_cvt to bf16 (compiler-selected)
    r.y = (__bf16)f.y;
    return r;
}

// ---------------------------------------------------------------------------
// Phase 1: projection GEMM  y = x @ W^T, cast to bf16, scatter to head layout.
//   x: [4096,1024] f32, W: [1024,1024] f32 (row-major)
//   transpose_v == 0 : out[b,h,l,hd]  (for Q and K)
//   transpose_v == 1 : out[b,h,hd,l]  (for V, pre-transposed for PV GEMM)
// Wave tile 64x64, block = 8 waves -> 256(M) x 128(N) per workgroup.
// ---------------------------------------------------------------------------
__global__ __launch_bounds__(256) void proj_gemm_kernel(
    const float* __restrict__ x, const float* __restrict__ W,
    __bf16* __restrict__ out, int transpose_v)
{
    const int lane = threadIdx.x & 31;
    const int wave = threadIdx.x >> 5;
    const int lx   = lane & 15;        // column / row-in-frag
    const int half = lane >> 4;        // lane half selects K sub-range
    const int m0 = blockIdx.y * 256 + (wave & 3) * 64;
    const int n0 = blockIdx.x * 128 + (wave >> 2) * 64;

    v8f acc[4][4];
#pragma unroll
    for (int i = 0; i < 4; ++i)
#pragma unroll
        for (int j = 0; j < 4; ++j) acc[i][j] = (v8f){};

    for (int k0 = 0; k0 < DIM; k0 += 32) {
        BFrag a[4], b[4];
        // A fragments (16x32 bf16): lane lx = row, VGPR v holds K pair
#pragma unroll
        for (int i = 0; i < 4; ++i) {
            const float* src = x + (size_t)(m0 + 16 * i + lx) * DIM;
#pragma unroll
            for (int v = 0; v < 8; ++v) {
                int k = k0 + ((v >> 2) << 4) + (half << 3) + ((v & 3) << 1);
                a[i].h[v] = cvt_pk_bf16(*(const float2*)(src + k));
            }
        }
        // B fragments (32x16 bf16): B[k,n] = W[n,k]; lane lx = col n
#pragma unroll
        for (int j = 0; j < 4; ++j) {
            const float* src = W + (size_t)(n0 + 16 * j + lx) * DIM;
#pragma unroll
            for (int v = 0; v < 8; ++v) {
                int k = k0 + ((v >> 2) << 4) + (half << 3) + ((v & 3) << 1);
                b[j].h[v] = cvt_pk_bf16(*(const float2*)(src + k));
            }
        }
#pragma unroll
        for (int i = 0; i < 4; ++i)
#pragma unroll
            for (int j = 0; j < 4; ++j)
                acc[i][j] = __builtin_amdgcn_wmma_f32_16x16x32_bf16(
                    false, a[i].v, false, b[j].v, (short)0, acc[i][j],
                    false, false);
    }

    // Scatter store: C frag element (r,lane): row m = r + 8*half, col = lx
#pragma unroll
    for (int i = 0; i < 4; ++i) {
#pragma unroll
        for (int j = 0; j < 4; ++j) {
            const int n = n0 + 16 * j + lx;
            const int h = n >> 6, hd = n & 63;
#pragma unroll
            for (int r = 0; r < 8; ++r) {
                const int m = m0 + 16 * i + r + (half << 3);
                const int bb = m >> 11, l = m & (SEQ - 1);
                size_t idx;
                if (!transpose_v)
                    idx = (((size_t)(bb * HEADS + h) * SEQ) + l) * HDIM + hd;
                else
                    idx = (((size_t)(bb * HEADS + h) * HDIM) + hd) * SEQ + l;
                out[idx] = (__bf16)acc[i][j][r];
            }
        }
    }
}

// ---------------------------------------------------------------------------
// Phase 2: flash attention with relative bias (q_idx - kv_idx), scale 1/8.
//   Qb,Kb: bf16 [B,H,L,HD] ; Vtb: bf16 [B,H,HD,L] ; out f32 [B,L,H*HD]
// One wave owns a 64-row q tile; 4 waves per workgroup; kv tiles of 32.
// ---------------------------------------------------------------------------
__global__ __launch_bounds__(128) void flash_attn_kernel(
    const __bf16* __restrict__ Qb,
    const __bf16* __restrict__ Kb,
    const __bf16* __restrict__ Vtb,
    float* __restrict__ out)
{
    const int lane = threadIdx.x & 31;
    const int wave = threadIdx.x >> 5;
    const int lx   = lane & 15;
    const int half = lane >> 4;

    const int bh = blockIdx.x;                 // [0, B*H)
    const int b  = bh >> 4, h = bh & 15;
    const int q0 = blockIdx.y * 256 + wave * 64;

    const __bf16* Q  = Qb  + (size_t)bh * SEQ * HDIM;
    const __bf16* K  = Kb  + (size_t)bh * SEQ * HDIM;
    const __bf16* Vt = Vtb + (size_t)bh * HDIM * SEQ;

    __shared__ __bf16 Pl[4][64][36];           // per-wave P staging (padded)

    // Preload Q A-fragments: qa[i][ks] covers q rows 16i..16i+15, d = 32ks..
    BFrag qa[4][2];
#pragma unroll
    for (int i = 0; i < 4; ++i) {
        const __bf16* src = Q + (size_t)(q0 + 16 * i + lx) * HDIM;
#pragma unroll
        for (int ks = 0; ks < 2; ++ks)
#pragma unroll
            for (int v = 0; v < 8; ++v) {
                int d = ks * 32 + ((v >> 2) << 4) + (half << 3) + ((v & 3) << 1);
                qa[i][ks].u[v] = *(const unsigned int*)(src + d);
            }
    }

    v8f oacc[4][4];
    float mrow[4][8], lrow[4][8];
#pragma unroll
    for (int i = 0; i < 4; ++i) {
#pragma unroll
        for (int jd = 0; jd < 4; ++jd) oacc[i][jd] = (v8f){};
#pragma unroll
        for (int r = 0; r < 8; ++r) { mrow[i][r] = -1.0e30f; lrow[i][r] = 0.0f; }
    }

    const float scale = 0.125f;                // HD^-0.5

    for (int kv0 = 0; kv0 < SEQ; kv0 += 32) {
        // prefetch next kv tile (K rows + Vt columns) into caches
        if (kv0 + 32 < SEQ) {
            __builtin_prefetch(K + (size_t)(kv0 + 32 + lane) * HDIM, 0, 3);
            __builtin_prefetch(Vt + (size_t)lane * SEQ + kv0 + 32, 0, 3);
        }

        // --- S = Q K^T : B frags from K rows (B[d,kv] = K[kv,d]) ---
        BFrag kb[2][2];
#pragma unroll
        for (int j = 0; j < 2; ++j) {
            const __bf16* src = K + (size_t)(kv0 + 16 * j + lx) * HDIM;
#pragma unroll
            for (int ks = 0; ks < 2; ++ks)
#pragma unroll
                for (int v = 0; v < 8; ++v) {
                    int d = ks * 32 + ((v >> 2) << 4) + (half << 3) + ((v & 3) << 1);
                    kb[j][ks].u[v] = *(const unsigned int*)(src + d);
                }
        }

        v8f s[4][2];
#pragma unroll
        for (int i = 0; i < 4; ++i)
#pragma unroll
            for (int j = 0; j < 2; ++j) {
                s[i][j] = (v8f){};
#pragma unroll
                for (int ks = 0; ks < 2; ++ks)
                    s[i][j] = __builtin_amdgcn_wmma_f32_16x16x32_bf16(
                        false, qa[i][ks].v, false, kb[j][ks].v, (short)0,
                        s[i][j], false, false);
            }

        // --- bias + online softmax ---
#pragma unroll
        for (int i = 0; i < 4; ++i) {
#pragma unroll
            for (int r = 0; r < 8; ++r) {
                const int qi = q0 + 16 * i + r + (half << 3);
#pragma unroll
                for (int j = 0; j < 2; ++j) {
                    const int kj = kv0 + 16 * j + lx;
                    s[i][j][r] = s[i][j][r] * scale + (float)(qi - kj);
                }
                float t = fmaxf(s[i][0][r], s[i][1][r]);
#pragma unroll
                for (int off = 1; off < 16; off <<= 1)
                    t = fmaxf(t, __shfl_xor(t, off, 32));
                const float mnew = fmaxf(mrow[i][r], t);
                const float corr = __expf(mrow[i][r] - mnew);
                mrow[i][r] = mnew;
                float psum = 0.0f;
#pragma unroll
                for (int j = 0; j < 2; ++j) {
                    const float p = __expf(s[i][j][r] - mnew);
                    s[i][j][r] = p;
                    psum += p;
                }
#pragma unroll
                for (int off = 1; off < 16; off <<= 1)
                    psum += __shfl_xor(psum, off, 32);
                lrow[i][r] = lrow[i][r] * corr + psum;
#pragma unroll
                for (int jd = 0; jd < 4; ++jd)
                    oacc[i][jd][r] *= corr;
            }
        }

        // --- P (C layout) -> LDS -> A fragments (wave-private, DS in-order) ---
#pragma unroll
        for (int i = 0; i < 4; ++i)
#pragma unroll
            for (int j = 0; j < 2; ++j)
#pragma unroll
                for (int r = 0; r < 8; ++r)
                    Pl[wave][16 * i + r + (half << 3)][16 * j + lx] =
                        (__bf16)s[i][j][r];

        BFrag pa[4];
#pragma unroll
        for (int i = 0; i < 4; ++i) {
            const __bf16* src = &Pl[wave][16 * i + lx][0];
#pragma unroll
            for (int v = 0; v < 8; ++v) {
                int k = ((v >> 2) << 4) + (half << 3) + ((v & 3) << 1);
                pa[i].u[v] = *(const unsigned int*)(src + k);
            }
        }

        // --- O += P V : B frags from Vt rows (B[kv,d] = Vt[d,kv]) ---
        BFrag bv[4];
#pragma unroll
        for (int jd = 0; jd < 4; ++jd) {
            const __bf16* src = Vt + (size_t)(16 * jd + lx) * SEQ;
#pragma unroll
            for (int v = 0; v < 8; ++v) {
                int kv = kv0 + ((v >> 2) << 4) + (half << 3) + ((v & 3) << 1);
                bv[jd].u[v] = *(const unsigned int*)(src + kv);
            }
        }
#pragma unroll
        for (int i = 0; i < 4; ++i)
#pragma unroll
            for (int jd = 0; jd < 4; ++jd)
                oacc[i][jd] = __builtin_amdgcn_wmma_f32_16x16x32_bf16(
                    false, pa[i].v, false, bv[jd].v, (short)0, oacc[i][jd],
                    false, false);
    }

    // --- finalize: O * (1/l), store f32 as [B, L, H*HD] ---
#pragma unroll
    for (int i = 0; i < 4; ++i)
#pragma unroll
        for (int r = 0; r < 8; ++r) {
            const float rinv = __frcp_rn(lrow[i][r]);
            const int l = q0 + 16 * i + r + (half << 3);
            float* dst = out + ((size_t)b * SEQ + l) * DIM + h * HDIM + lx;
#pragma unroll
            for (int jd = 0; jd < 4; ++jd)
                dst[16 * jd] = oacc[i][jd][r] * rinv;
        }
}

// ---------------------------------------------------------------------------
extern "C" void kernel_launch(void* const* d_in, const int* in_sizes, int n_in,
                              void* d_out, int out_size, void* d_ws,
                              size_t ws_size, hipStream_t stream) {
    (void)in_sizes; (void)n_in; (void)out_size; (void)ws_size;
    const float* q  = (const float*)d_in[0];
    const float* k  = (const float*)d_in[1];
    const float* v  = (const float*)d_in[2];
    const float* Wq = (const float*)d_in[3];
    const float* Wk = (const float*)d_in[4];
    const float* Wv = (const float*)d_in[5];

    const size_t elems = (size_t)BATCH * HEADS * SEQ * HDIM;   // 4M bf16 each
    __bf16* Qb  = (__bf16*)d_ws;
    __bf16* Kb  = Qb + elems;
    __bf16* Vtb = Kb + elems;

    dim3 pblk(256);
    dim3 pgrd(DIM / 128, ROWS / 256, 1);   // (8, 16)
    proj_gemm_kernel<<<pgrd, pblk, 0, stream>>>(q, Wq, Qb, 0);
    proj_gemm_kernel<<<pgrd, pblk, 0, stream>>>(k, Wk, Kb, 0);
    proj_gemm_kernel<<<pgrd, pblk, 0, stream>>>(v, Wv, Vtb, 1);

    dim3 ablk(128);
    dim3 agrd(BATCH * HEADS, SEQ / 256, 1); // (32, 8)
    flash_attn_kernel<<<agrd, ablk, 0, stream>>>(Qb, Kb, Vtb, (float*)d_out);
}
